// SANet_14577119003254
// MI455X (gfx1250) — compile-verified, matrix operand order checked by reading, not run on hardware
//
#include <hip/hip_runtime.h>

// SANet attention for MI455X (gfx1250), wave32 + WMMA f16 (f32 accum).
//
// Roofline: ~172 GFLOP total, dominated by the two 4096x4096x512 batched
// attention GEMMs -> compute-bound; all GEMMs run on v_wmma_f32_16x16x32_f16.
// Mean-var norm is folded into the projection weights (per batch), so every
// GEMM consumes raw f16 operands and every fragment is two 16B vector loads.
// S (134MB f16) is L2-resident (192MB), so the two-pass softmax is nearly free.
// Workspace requirement: ~240 MB.

typedef __attribute__((ext_vector_type(16))) _Float16 v16h;
typedef __attribute__((ext_vector_type(8)))  _Float16 v8h;
typedef __attribute__((ext_vector_type(8)))  float    v8f;

#define BN  4
#define CH  512
#define HW  4096
#define EPSN 1e-5f

union Frag16 { v16h v; v8h half8[2]; };

// f16 16x32 A/B fragment: per lane, elements 0..7 -> K = h*8+0..7,
// elements 8..15 -> K = 16+h*8+0..7 (ISA 7.12.2) => two contiguous 8-elt runs.
__device__ __forceinline__ v16h load_frag_f16(const _Float16* __restrict__ base) {
    Frag16 f;
    f.half8[0] = *(const v8h*)(base);
    f.half8[1] = *(const v8h*)(base + 16);
    return f.v;
}

// ---------------------------------------------------------------------------
// Kernel 1: per-(b,c) spatial mean / rstd over 4096 pixels
// ---------------------------------------------------------------------------
__global__ void stats_kernel(const float* __restrict__ x,
                             float* __restrict__ mean, float* __restrict__ rstd) {
    __shared__ float s1[256], s2[256];
    const int row = blockIdx.x;                       // b*CH + c
    const float* p = x + (size_t)row * HW;
    float sum = 0.f, sq = 0.f;
    for (int i = threadIdx.x; i < HW; i += 256) { float v = p[i]; sum += v; sq += v * v; }
    s1[threadIdx.x] = sum; s2[threadIdx.x] = sq;
    __syncthreads();
    for (int off = 128; off > 0; off >>= 1) {
        if ((int)threadIdx.x < off) {
            s1[threadIdx.x] += s1[threadIdx.x + off];
            s2[threadIdx.x] += s2[threadIdx.x + off];
        }
        __syncthreads();
    }
    if (threadIdx.x == 0) {
        float m = s1[0] * (1.0f / HW);
        float var = s2[0] * (1.0f / HW) - m * m;
        mean[row] = m;
        rstd[row] = rsqrtf(var + EPSN);
    }
}

// ---------------------------------------------------------------------------
// Kernel 2: transpose + convert: X f32 [B,CH,HW] -> Xt f16 [B,HW,CH].
// 64x64 LDS tile; reads and writes both coalesced; 66-elt pitch kills
// bank conflicts on the transposed read.
// ---------------------------------------------------------------------------
__global__ void transpose_f16_kernel(const float* __restrict__ X,
                                     _Float16* __restrict__ Xt) {
    __shared__ _Float16 tile[64][66];
    const int b  = blockIdx.z;
    const int c0 = blockIdx.y * 64;
    const int p0 = blockIdx.x * 64;                   // hw base
    const int tx = threadIdx.x & 63;
    const int ty = threadIdx.x >> 6;                  // 0..3
    const float* Xb = X + ((size_t)b * CH + c0) * HW + p0;
#pragma unroll
    for (int i = 0; i < 16; ++i) {
        const int c = ty + i * 4;
        tile[c][tx] = (_Float16)Xb[(size_t)c * HW + tx];   // coalesced along hw
    }
    __syncthreads();
    _Float16* Ob = Xt + ((size_t)b * HW + p0) * CH + c0;
#pragma unroll
    for (int i = 0; i < 16; ++i) {
        const int p = ty + i * 4;
        Ob[(size_t)p * CH + tx] = tile[tx][p];             // coalesced along c
    }
}

// ---------------------------------------------------------------------------
// Kernel 3: fold mean-var norm into conv weights (per batch):
//   Wh[b][o][c] = W[o][c] * rstd[b][c]          (f16)
//   biasf[b][o] = bias[o] - sum_c Wh*mean[b][c] (f32)
// With null stats this is just an f32->f16 weight convert + bias copy.
// ---------------------------------------------------------------------------
__global__ void fold_kernel(const float* __restrict__ W,     // [CH, CH]
                            const float* __restrict__ bias,  // [CH]
                            const float* __restrict__ mean,  // [nb*CH] or null
                            const float* __restrict__ rstd,  // [nb*CH] or null
                            _Float16* __restrict__ Wh,       // [nb, CH, CH]
                            float* __restrict__ biasf)       // [nb, CH]
{
    __shared__ float red[256];
    const int co = blockIdx.x, b = blockIdx.y;
    float partial = 0.f;
    for (int ci = threadIdx.x; ci < CH; ci += 256) {
        float w = W[(size_t)co * CH + ci];
        if (rstd) w *= rstd[b * CH + ci];
        Wh[((size_t)b * CH + co) * CH + ci] = (_Float16)w;
        if (mean) partial += w * mean[b * CH + ci];
    }
    red[threadIdx.x] = partial;
    __syncthreads();
    for (int off = 128; off > 0; off >>= 1) {
        if ((int)threadIdx.x < off) red[threadIdx.x] += red[threadIdx.x + off];
        __syncthreads();
    }
    if (threadIdx.x == 0) biasf[b * CH + co] = bias[co] - red[0];
}

// ---------------------------------------------------------------------------
// Kernel 4: pure f16 GEMM projection: Out[b][hw][co] = Xt[b][hw][:] . Wh[b][co][:]
// Wave computes a 16m x 64n tile (4 accumulators); A fragment loaded once per
// K-step (2x16B) and reused across 4 WMMAs; W fragments 2x16B each.
// transposeOut=1 writes Out as [B, CH, HW] with packed 16B v8h stores (for V).
// ---------------------------------------------------------------------------
__global__ void __launch_bounds__(256) proj_kernel(
        const _Float16* __restrict__ Xt,     // [B, HW, CH]
        const _Float16* __restrict__ Wh,     // [nb, CH, CH]
        const float* __restrict__ biasf,     // [nb, CH]
        _Float16* __restrict__ Out,          // [B, HW, CH] or [B, CH, HW]
        int perBatch, int transposeOut)
{
    const int lane = threadIdx.x & 31;
    const int wv   = threadIdx.x >> 5;
    const int h    = lane >> 4;
    const int ln   = lane & 15;
    const int b    = blockIdx.y;
    const int n0   = blockIdx.x * 64;                 // c_out group base
    const int bw   = perBatch ? b : 0;

    const _Float16* Xb = Xt + (size_t)b * HW * CH;
    const _Float16* Wb = Wh + (size_t)bw * CH * CH;

    float bv[4];
#pragma unroll
    for (int j = 0; j < 4; ++j) bv[j] = biasf[bw * CH + n0 + j * 16 + ln];

    for (int mt = wv; mt < HW / 16; mt += 8) {
        const int m0 = mt * 16;
        v8f acc[4];
#pragma unroll
        for (int j = 0; j < 4; ++j)
#pragma unroll
            for (int r = 0; r < 8; ++r) acc[j][r] = bv[j];   // bias on column n

#pragma unroll 1
        for (int s = 0; s < 16; ++s) {
            v16h af = load_frag_f16(Xb + (size_t)(m0 + ln) * CH + s * 32 + h * 8);
#pragma unroll
            for (int j = 0; j < 4; ++j) {
                v16h wf = load_frag_f16(Wb + (size_t)(n0 + j * 16 + ln) * CH + s * 32 + h * 8);
                acc[j] = __builtin_amdgcn_wmma_f32_16x16x32_f16(false, af, false, wf,
                                                                (short)0, acc[j], false, false);
            }
        }

        if (transposeOut) {
            // Out[b][c][hw]: each lane's 8 rows are contiguous in m -> 16B store.
#pragma unroll
            for (int j = 0; j < 4; ++j) {
                v8h pk;
#pragma unroll
                for (int r = 0; r < 8; ++r) pk[r] = (_Float16)acc[j][r];
                _Float16* dst = Out + ((size_t)b * CH + n0 + j * 16 + ln) * HW + m0 + 8 * h;
                *(v8h*)dst = pk;
            }
        } else {
#pragma unroll
            for (int j = 0; j < 4; ++j)
#pragma unroll
                for (int r = 0; r < 8; ++r) {
                    const int m = m0 + r + 8 * h;
                    Out[((size_t)b * HW + m) * CH + n0 + j * 16 + ln] = (_Float16)acc[j][r];
                }
        }
    }
}

// ---------------------------------------------------------------------------
// Kernel 5: S = Q * K^T (pure WMMA GEMM, f16 store).
// One wave owns a 16-query tile; Q fragments (128 VGPRs) cached for K=512.
// launch_bounds(,1): allow full register file at 1 wave/SIMD, no spills.
// ---------------------------------------------------------------------------
__global__ void __launch_bounds__(256, 1) qk_kernel(
        const _Float16* __restrict__ Q,   // [B, HW, CH]
        const _Float16* __restrict__ K,   // [B, HW, CH]
        _Float16* __restrict__ S)         // [B, HW, HW]
{
    const int lane = threadIdx.x & 31;
    const int wv   = threadIdx.x >> 5;
    const int h    = lane >> 4;
    const int ln   = lane & 15;
    const int b    = blockIdx.y;
    const int q0   = (blockIdx.x * 8 + wv) * 16;

    const _Float16* Qb = Q + (size_t)b * HW * CH;
    const _Float16* Kb = K + (size_t)b * HW * CH;
    _Float16* Sb = S + (size_t)b * HW * HW;

    // Cache all Q A-fragments for K=512 (16 x v16h = 128 VGPRs).
    v16h qf[16];
#pragma unroll
    for (int s = 0; s < 16; ++s)
        qf[s] = load_frag_f16(Qb + (size_t)(q0 + ln) * CH + s * 32 + h * 8);

    for (int kt = 0; kt < HW / 16; ++kt) {
        const int k0 = kt * 16;
        v8f acc;
#pragma unroll
        for (int r = 0; r < 8; ++r) acc[r] = 0.f;

#pragma unroll
        for (int s = 0; s < 16; ++s) {
            v16h kf = load_frag_f16(Kb + (size_t)(k0 + ln) * CH + s * 32 + h * 8);
            acc = __builtin_amdgcn_wmma_f32_16x16x32_f16(false, qf[s], false, kf,
                                                         (short)0, acc, false, false);
        }
#pragma unroll
        for (int r = 0; r < 8; ++r)
            Sb[(size_t)(q0 + r + 8 * h) * HW + k0 + ln] = (_Float16)acc[r];
    }
}

// ---------------------------------------------------------------------------
// Kernel 6: per-row softmax stats over S (L2-resident): rowM = max, rowL = sum
// of exp(s-max). Online (m,l) merge; vectorized v8h reads, one block per row.
// ---------------------------------------------------------------------------
__global__ void softmax_stats_kernel(const _Float16* __restrict__ S,
                                     float* __restrict__ rowM,
                                     float* __restrict__ rowL) {
    __shared__ float sm[256], sl[256];
    const v8h* row = (const v8h*)(S + (size_t)blockIdx.x * HW);
    float m = -1e30f, l = 0.f;
    for (int c = threadIdx.x; c < HW / 8; c += 256) {
        v8h v = row[c];
#pragma unroll
        for (int e = 0; e < 8; ++e) {
            const float x = (float)v[e];
            if (x > m) { l *= __expf(m - x); m = x; }
            l += __expf(x - m);
        }
    }
    sm[threadIdx.x] = m; sl[threadIdx.x] = l;
    __syncthreads();
    for (int off = 128; off > 0; off >>= 1) {
        if ((int)threadIdx.x < off) {
            const float m2 = sm[threadIdx.x + off], l2 = sl[threadIdx.x + off];
            const float m1 = sm[threadIdx.x],       l1 = sl[threadIdx.x];
            const float nm = fmaxf(m1, m2);
            sm[threadIdx.x] = nm;
            sl[threadIdx.x] = l1 * __expf(m1 - nm) + l2 * __expf(m2 - nm);
        }
        __syncthreads();
    }
    if (threadIdx.x == 0) { rowM[blockIdx.x] = sm[0]; rowL[blockIdx.x] = sl[0]; }
}

// ---------------------------------------------------------------------------
// Kernel 7: AO = softmax(S) * V.  P rebuilt on the fly from f16 S + row stats.
// V is transposed [B, CH, HW] so every fragment is two 16B vector loads.
// Block = (b, 16-query tile); wave w owns 64 output channels (4 c-tiles).
// ---------------------------------------------------------------------------
__global__ void __launch_bounds__(256) pv_kernel(
        const _Float16* __restrict__ S,   // [B, HW, HW]
        const float* __restrict__ rowM,
        const float* __restrict__ rowL,
        const _Float16* __restrict__ Vt,  // [B, CH, HW]  (transposed values)
        _Float16* __restrict__ AO)        // [B, HW, CH]
{
    const int lane = threadIdx.x & 31;
    const int wv   = threadIdx.x >> 5;
    const int h    = lane >> 4;
    const int ln   = lane & 15;
    const int b    = blockIdx.y;
    const int q0   = blockIdx.x * 16;

    const _Float16* Sb = S  + (size_t)b * HW * HW;
    const _Float16* Vb = Vt + (size_t)b * CH * HW;

    const int qrow = q0 + ln;                          // A-fragment row for this lane
    const float Mr   = rowM[b * HW + qrow];
    const float invL = 1.0f / rowL[b * HW + qrow];

    v8f acc[4];
#pragma unroll
    for (int j = 0; j < 4; ++j)
#pragma unroll
        for (int r = 0; r < 8; ++r) acc[j][r] = 0.f;

#pragma unroll 1
    for (int ks = 0; ks < HW / 32; ++ks) {
        const int kb = ks * 32;
        // P A-fragment: two 16B loads of raw S, then exp()/l in f32.
        const _Float16* sbase = Sb + (size_t)qrow * HW + kb + h * 8;
        Frag16 sf;
        sf.half8[0] = *(const v8h*)(sbase);
        sf.half8[1] = *(const v8h*)(sbase + 16);
        if (ks + 1 < HW / 32) __builtin_prefetch(sbase + 32, 0, 1);
        v16h pf;
#pragma unroll
        for (int e = 0; e < 16; ++e)
            pf[e] = (_Float16)(__expf((float)sf.v[e] - Mr) * invL);

#pragma unroll
        for (int j = 0; j < 4; ++j) {
            const int c = (wv * 4 + j) * 16 + ln;      // B-fragment column
            v16h vf = load_frag_f16(Vb + (size_t)c * HW + kb + h * 8);
            acc[j] = __builtin_amdgcn_wmma_f32_16x16x32_f16(false, pf, false, vf,
                                                            (short)0, acc[j], false, false);
        }
    }
#pragma unroll
    for (int j = 0; j < 4; ++j) {
        const int c0 = (wv * 4 + j) * 16;
#pragma unroll
        for (int r = 0; r < 8; ++r)
            AO[((size_t)b * HW + q0 + r + 8 * h) * CH + c0 + ln] = (_Float16)acc[j][r];
    }
}

// ---------------------------------------------------------------------------
// Kernel 8: out[b][co][q] = sum_c Wo16[co][c] * AO[b][q][c] + bo[co] + content
// M = co, N = q. Wave computes 64co x 16q (4 accumulators); the AO B-fragment
// is loaded once per K-step and reused across 4 WMMAs.
// ---------------------------------------------------------------------------
__global__ void __launch_bounds__(256) out_kernel(
        const _Float16* __restrict__ AO,   // [B, HW, CH]
        const _Float16* __restrict__ Wo16, // [CH, CH]
        const float* __restrict__ bo,      // [CH]
        const float* __restrict__ content, // [B, CH, HW]
        float* __restrict__ out)           // [B, CH, HW]
{
    const int lane = threadIdx.x & 31;
    const int wv   = threadIdx.x >> 5;
    const int h    = lane >> 4;
    const int ln   = lane & 15;
    const int b    = blockIdx.y;
    const int co0  = blockIdx.x * 64;                  // output-channel group

    float biasv[4][8];
#pragma unroll
    for (int j = 0; j < 4; ++j)
#pragma unroll
        for (int r = 0; r < 8; ++r) biasv[j][r] = bo[co0 + j * 16 + r + 8 * h];

    const _Float16* Ab = AO + (size_t)b * HW * CH;

    for (int qt = wv; qt < HW / 16; qt += 8) {
        const int q0 = qt * 16;
        v8f acc[4];
#pragma unroll
        for (int j = 0; j < 4; ++j)
#pragma unroll
            for (int r = 0; r < 8; ++r) acc[j][r] = biasv[j][r];

#pragma unroll 1
        for (int s = 0; s < 16; ++s) {
            // B fragment from AO: n = q, K = c contiguous per lane.
            v16h bf = load_frag_f16(Ab + (size_t)(q0 + ln) * CH + s * 32 + h * 8);
#pragma unroll
            for (int j = 0; j < 4; ++j) {
                v16h af = load_frag_f16(Wo16 + (size_t)(co0 + j * 16 + ln) * CH + s * 32 + h * 8);
                acc[j] = __builtin_amdgcn_wmma_f32_16x16x32_f16(false, af, false, bf,
                                                                (short)0, acc[j], false, false);
            }
        }
#pragma unroll
        for (int j = 0; j < 4; ++j)
#pragma unroll
            for (int r = 0; r < 8; ++r) {
                const size_t idx = ((size_t)b * CH + co0 + j * 16 + r + 8 * h) * HW + q0 + ln;
                out[idx] = acc[j][r] + content[idx];   // coalesced across lanes
            }
    }
}

// ---------------------------------------------------------------------------
extern "C" void kernel_launch(void* const* d_in, const int* in_sizes, int n_in,
                              void* d_out, int out_size, void* d_ws, size_t ws_size,
                              hipStream_t stream)
{
    const float* content = (const float*)d_in[0];
    const float* style   = (const float*)d_in[1];
    const float* f_w = (const float*)d_in[2];
    const float* f_b = (const float*)d_in[3];
    const float* g_w = (const float*)d_in[4];
    const float* g_b = (const float*)d_in[5];
    const float* h_w = (const float*)d_in[6];
    const float* h_b = (const float*)d_in[7];
    const float* o_w = (const float*)d_in[8];
    const float* o_b = (const float*)d_in[9];
    float* out = (float*)d_out;

    char* ws = (char*)d_ws;
    size_t off = 0;
    auto take = [&](size_t bytes) -> char* {
        char* p = ws + off;
        off = (off + bytes + 255) & ~(size_t)255;
        return p;
    };
    float*    meanC = (float*)take((size_t)BN * CH * sizeof(float));
    float*    rstdC = (float*)take((size_t)BN * CH * sizeof(float));
    float*    meanS = (float*)take((size_t)BN * CH * sizeof(float));
    float*    rstdS = (float*)take((size_t)BN * CH * sizeof(float));
    _Float16* Xc    = (_Float16*)take((size_t)BN * HW * CH * sizeof(_Float16)); // content f16 [B,HW,CH]
    _Float16* Xs    = (_Float16*)take((size_t)BN * HW * CH * sizeof(_Float16)); // style   f16 [B,HW,CH]
    _Float16* Wf16  = (_Float16*)take((size_t)BN * CH * CH * sizeof(_Float16)); // folded f weights
    _Float16* Wg16  = (_Float16*)take((size_t)BN * CH * CH * sizeof(_Float16)); // folded g weights
    _Float16* Wh16  = (_Float16*)take((size_t)CH * CH * sizeof(_Float16));      // h weights (f16)
    _Float16* Wo16  = (_Float16*)take((size_t)CH * CH * sizeof(_Float16));      // o weights (f16)
    float*    bf16f = (float*)take((size_t)BN * CH * sizeof(float));
    float*    bg16f = (float*)take((size_t)BN * CH * sizeof(float));
    float*    bh16f = (float*)take((size_t)CH * sizeof(float));
    float*    bo16f = (float*)take((size_t)CH * sizeof(float));
    _Float16* Qf    = (_Float16*)take((size_t)BN * HW * CH * sizeof(_Float16));
    _Float16* Kf    = (_Float16*)take((size_t)BN * HW * CH * sizeof(_Float16));
    _Float16* Vt    = (_Float16*)take((size_t)BN * HW * CH * sizeof(_Float16));
    _Float16* Sm    = (_Float16*)take((size_t)BN * HW * HW * sizeof(_Float16));
    float*    rowM  = (float*)take((size_t)BN * HW * sizeof(float));
    float*    rowL  = (float*)take((size_t)BN * HW * sizeof(float));
    _Float16* AO    = (_Float16*)take((size_t)BN * HW * CH * sizeof(_Float16));
    (void)ws_size; (void)in_sizes; (void)n_in; (void)out_size;

    // 1) normalization stats
    stats_kernel<<<dim3(BN * CH), 256, 0, stream>>>(content, meanC, rstdC);
    stats_kernel<<<dim3(BN * CH), 256, 0, stream>>>(style,   meanS, rstdS);

    // 2) raw inputs -> f16, transposed to [B, HW, CH]
    transpose_f16_kernel<<<dim3(HW / 64, CH / 64, BN), 256, 0, stream>>>(content, Xc);
    transpose_f16_kernel<<<dim3(HW / 64, CH / 64, BN), 256, 0, stream>>>(style,   Xs);

    // 3) fold mean-var norm into f/g weights (per batch); plain f16 convert h/o
    fold_kernel<<<dim3(CH, BN), 256, 0, stream>>>(f_w, f_b, meanC, rstdC, Wf16, bf16f);
    fold_kernel<<<dim3(CH, BN), 256, 0, stream>>>(g_w, g_b, meanS, rstdS, Wg16, bg16f);
    fold_kernel<<<dim3(CH, 1),  256, 0, stream>>>(h_w, h_b, nullptr, nullptr, Wh16, bh16f);
    fold_kernel<<<dim3(CH, 1),  256, 0, stream>>>(o_w, o_b, nullptr, nullptr, Wo16, bo16f);

    // 4) pure f16 GEMM projections (V transposed for the PV pass)
    proj_kernel<<<dim3(CH / 64, BN), 256, 0, stream>>>(Xc, Wf16, bf16f, Qf, 1, 0);
    proj_kernel<<<dim3(CH / 64, BN), 256, 0, stream>>>(Xs, Wg16, bg16f, Kf, 1, 0);
    proj_kernel<<<dim3(CH / 64, BN), 256, 0, stream>>>(Xs, Wh16, bh16f, Vt, 0, 1);

    // 5) S = Q K^T  (f16, L2-resident)
    qk_kernel<<<dim3(HW / 16 / 8, BN), 256, 0, stream>>>(Qf, Kf, Sm);

    // 6) per-row softmax stats
    softmax_stats_kernel<<<dim3(BN * HW), 256, 0, stream>>>(Sm, rowM, rowL);

    // 7) AO = softmax(S) V
    pv_kernel<<<dim3(HW / 16, BN), 256, 0, stream>>>(Sm, rowM, rowL, Vt, AO);

    // 8) final 1x1 conv + residual
    out_kernel<<<dim3(CH / 64, BN), 256, 0, stream>>>(AO, Wo16, bo16f, content, out);
}
